// NeuralLV_29575144801187
// MI455X (gfx1250) — matrix operand order repaired and checked by reading, not compiled, not run on hardware
//
#include <hip/hip_runtime.h>
#include <cmath>

// ---------------------------------------------------------------------------
// Neural local-vol Monte Carlo for MI455X (gfx1250, wave32, WMMA).
// Fused pipeline per step per net:
//   k_gemm1_fused : layer1(analytic BN from S-stats, in LDS) -> WMMA(W2)
//                   -> y2 f32 [B,128] + per-block feature sum/sumsq partials
//   k_statsfin    : deterministic partial reduction -> per-feature mean/var
//   lev:   k_lev_final  (BN+lrelu inline, 1x128 dot, softplus, path update)
//   hedge: k_gemm96_acc (BN+lrelu inline staging -> WMMA(W3h) -> fused
//                        hedging[f][b] += coef_b * (y + b3_f) RMW)
// All reductions are fixed-shape tree reductions (bitwise deterministic).
// ---------------------------------------------------------------------------

#define B_PATHS   32768
#define LAYER     128
#define N_MAT     4
#define N_STRIKES 21
#define HROWS     84          // N_STRIKES * N_MAT
#define HROWS_PAD 96          // padded to 6 WMMA m-tiles
#define TP        136         // LDS tile row stride in halves (272B -> 68 dword-banks)
#define NBLK1     (B_PATHS / 128)   // 256 blocks for k_gemm1_fused
#define SQDT_F    0.17677669529663687f

typedef __attribute__((ext_vector_type(16))) _Float16 v16h;
typedef __attribute__((ext_vector_type(8)))  _Float16 v8h;
typedef __attribute__((ext_vector_type(8)))  float    v8f;

__device__ __forceinline__ float lrelu(float x) { return x > 0.f ? x : 0.01f * x; }

// --- one-time weight conversion: W2 (both nets) f32->f16; hedge W3 f32->f16 padded 84->96 rows
__global__ void k_convert(const float* __restrict__ W2l, const float* __restrict__ W2h,
                          const float* __restrict__ W3h,
                          _Float16* __restrict__ w2l16, _Float16* __restrict__ w2h16,
                          _Float16* __restrict__ w3h16) {
  int i = blockIdx.x * blockDim.x + threadIdx.x;      // 0 .. 65535
  w2l16[i] = (_Float16)W2l[i];
  w2h16[i] = (_Float16)W2h[i];
  if (i < N_MAT * HROWS_PAD * LAYER) {
    int mat = i / (HROWS_PAD * LAYER);
    int rem = i % (HROWS_PAD * LAYER);
    int r = rem / LAYER, c = rem % LAYER;
    float v = (r < HROWS) ? W3h[(mat * HROWS + r) * LAYER + c] : 0.f;
    w3h16[i] = (_Float16)v;
  }
}

__global__ void k_init(float* __restrict__ hedging, float* __restrict__ S) {
  int i = blockIdx.x * blockDim.x + threadIdx.x;
  if (i < HROWS * B_PATHS) hedging[i] = 0.f;
  if (i < B_PATHS)         S[i] = 1.0f;              // S0
}

// --- scalar mean/var of S (single block, deterministic tree reduce)
__global__ __launch_bounds__(1024) void k_stats_S(const float* __restrict__ S,
                                                  float* __restrict__ st) {
  __shared__ float ss[1024], sq[1024];
  float s = 0.f, q = 0.f;
  for (int i = threadIdx.x; i < B_PATHS; i += 1024) { float v = S[i]; s += v; q += v * v; }
  ss[threadIdx.x] = s; sq[threadIdx.x] = q;
  __syncthreads();
  for (int o = 512; o > 0; o >>= 1) {
    if ((int)threadIdx.x < o) { ss[threadIdx.x] += ss[threadIdx.x + o]; sq[threadIdx.x] += sq[threadIdx.x + o]; }
    __syncthreads();
  }
  if (threadIdx.x == 0) { float m = ss[0] / B_PATHS; st[0] = m; st[1] = sq[0] / B_PATHS - m * m; }
}

// --- load a contiguous-k A fragment (16x32 f16, k = 2*aoff + s) for row m, k-chunk kc
__device__ __forceinline__ v16h load_afrag(const _Float16* __restrict__ W, int m, int kc, int aoff) {
  v8h lo = *(const v8h*)(W + m * LAYER + kc * 32 + 2 * aoff);
  v8h hh = *(const v8h*)(W + m * LAYER + kc * 32 + 2 * aoff + 8);
  v16h a;
#pragma unroll
  for (int e = 0; e < 8; ++e) { a[e] = lo[e]; a[8 + e] = hh[e]; }
  return a;
}

// --- fused: layer1 (analytic BN: y = A_j*(S-meanS)+be_j, lrelu) staged in LDS,
//     WMMA with W2, y2 f32 out, per-block feature sum/sumsq partials out.
__global__ __launch_bounds__(256) void k_gemm1_fused(
    const _Float16* __restrict__ W2,      // [128][128] f16, maturity pre-offset
    const float* __restrict__ W1,         // [N_MAT][128][2]
    const float* __restrict__ g1, const float* __restrict__ be1, int idx,
    const float* __restrict__ S, const float* __restrict__ st,
    float* __restrict__ y2,               // [B,128]
    float* __restrict__ partials) {       // [NBLK1][2][128]
  __shared__ _Float16 tile[128 * TP];     // 34 KB; reused as f32[4096] for reduction
  __shared__ float sA[LAYER], sBe[LAYER], sS[128];

  int bb0 = blockIdx.x * 128;
  float meanS = st[0], varS = st[1];
  if (threadIdx.x < 128) {
    int j = threadIdx.x;
    float w1 = W1[(idx * LAYER + j) * 2 + 1];
    sA[j]  = g1[idx * LAYER + j] * w1 * rsqrtf(w1 * w1 * varS + 1e-5f);
    sBe[j] = be1[idx * LAYER + j];
    sS[j]  = S[bb0 + j] - meanS;
  }
  __syncthreads();

  // stage layer1 output tile (f16, padded rows)
  for (int c = threadIdx.x; c < 2048; c += 256) {
    int row = c >> 4, j0 = (c & 15) * 8;
    float sd = sS[row];
    v8h o;
#pragma unroll
    for (int e = 0; e < 8; ++e)
      o[e] = (_Float16)lrelu(sA[j0 + e] * sd + sBe[j0 + e]);
    *(v8h*)(tile + row * TP + j0) = o;
  }
  __syncthreads();

  int wave = threadIdx.x >> 5;
  int lane = threadIdx.x & 31;
  int hi   = lane >> 4;
  int m0   = wave * 16;
  int m    = m0 + (lane & 15);
  int aoff = hi * 8;

  v16h A[4];
#pragma unroll
  for (int kc = 0; kc < 4; ++kc) A[kc] = load_afrag(W2, m, kc, aoff);

  int n = lane & 15;
  int koff = hi * 16;

  float ls[8] = {}, lq[8] = {};
#pragma unroll
  for (int nt = 0; nt < 8; ++nt) {
    v8f acc = {};
#pragma unroll
    for (int kc = 0; kc < 4; ++kc) {
      v16h Bf = *(const v16h*)(tile + (nt * 16 + n) * TP + kc * 32 + koff);
      acc = __builtin_amdgcn_wmma_f32_16x16x32_f16(false, A[kc], false, Bf,
                                                   (short)0, acc, false, false);
    }
    float* dst = y2 + (size_t)(bb0 + nt * 16 + n) * LAYER + m0 + hi * 8;
#pragma unroll
    for (int r = 0; r < 8; ++r) {
      float v = acc[r];
      dst[r] = v;
      ls[r] += v; lq[r] += v * v;
    }
  }

  // per-block feature partials: reduce over the 16 n-lanes (reuse tile LDS)
  __syncthreads();
  float* redS = (float*)tile;         // [128][16]
  float* redQ = redS + 2048;          // [128][16]
#pragma unroll
  for (int r = 0; r < 8; ++r) {
    int f = m0 + hi * 8 + r;
    redS[f * 16 + n] = ls[r];
    redQ[f * 16 + n] = lq[r];
  }
  __syncthreads();
  if (threadIdx.x < 128) {
    int f = threadIdx.x;
    float s = 0.f, q = 0.f;
#pragma unroll
    for (int k = 0; k < 16; ++k) { s += redS[f * 16 + k]; q += redQ[f * 16 + k]; }
    partials[(size_t)blockIdx.x * 256 + f]       = s;
    partials[(size_t)blockIdx.x * 256 + 128 + f] = q;
  }
}

// --- finish feature stats: fs[2j]=mean, fs[2j+1]=biased var (128 blocks)
__global__ __launch_bounds__(128) void k_statsfin(const float* __restrict__ P,
                                                  float* __restrict__ fs) {
  int j = blockIdx.x;
  __shared__ float ss[128], sq[128];
  float s = 0.f, q = 0.f;
  for (int bi = threadIdx.x; bi < NBLK1; bi += 128) {
    s += P[(size_t)bi * 256 + j];
    q += P[(size_t)bi * 256 + 128 + j];
  }
  ss[threadIdx.x] = s; sq[threadIdx.x] = q;
  __syncthreads();
  for (int o = 64; o > 0; o >>= 1) {
    if ((int)threadIdx.x < o) { ss[threadIdx.x] += ss[threadIdx.x + o]; sq[threadIdx.x] += sq[threadIdx.x + o]; }
    __syncthreads();
  }
  if (threadIdx.x == 0) {
    float m = ss[0] / B_PATHS;
    fs[2 * j] = m;
    fs[2 * j + 1] = sq[0] / B_PATHS - m * m;
  }
}

// --- lev head: BN+lrelu inline, 1x128 dot, softplus, path update
__global__ __launch_bounds__(256) void k_lev_final(
    const float* __restrict__ y2, const float* __restrict__ fs,
    const float* __restrict__ g2, const float* __restrict__ be2,
    const float* __restrict__ W3, const float* __restrict__ b3, int idx,
    const float* __restrict__ S_old, const float* __restrict__ nrm,
    float* __restrict__ S_new, float* __restrict__ dif) {
  __shared__ float sc1[LAYER], sc0[LAYER], sw[LAYER];
  if (threadIdx.x < 128) {
    int j = threadIdx.x;
    float c1 = g2[idx * LAYER + j] * rsqrtf(fs[2 * j + 1] + 1e-5f);
    sc1[j] = c1;
    sc0[j] = be2[idx * LAYER + j] - c1 * fs[2 * j];
    sw[j]  = W3[idx * LAYER + j];
  }
  __syncthreads();
  int b = blockIdx.x * blockDim.x + threadIdx.x;
  const float* row = y2 + (size_t)b * LAYER;
  float acc = b3[idx];
#pragma unroll 8
  for (int k = 0; k < LAYER; ++k)
    acc += sw[k] * lrelu(sc1[k] * row[k] + sc0[k]);
  float sp = fmaxf(acc, 0.f) + log1pf(expf(-fabsf(acc)));   // softplus
  const float RFR = 0.05f, DT = 1.f / 32.f;
  float s = S_old[b];
  float drift = s * RFR / (1.f + fabsf(s * RFR) * SQDT_F);
  float d = sp / (1.f + sp * SQDT_F);                       // sp >= 0
  S_new[b] = s + drift * DT + d * SQDT_F * nrm[b];
  dif[b] = d;
}

// --- hedge head fused: BN+lrelu while staging, WMMA(W3h), fused hedging RMW
__global__ __launch_bounds__(192) void k_gemm96_acc(
    const _Float16* __restrict__ W,       // [96][128] f16, maturity pre-offset
    const float* __restrict__ y2, const float* __restrict__ fs,
    const float* __restrict__ g2, const float* __restrict__ be2,
    const float* __restrict__ b3h, int idx,
    const float* __restrict__ S_old, const float* __restrict__ dif,
    const float* __restrict__ nrm, float dfm1,
    float* __restrict__ hedging) {
  __shared__ _Float16 tile[64 * TP];      // 17 KB
  __shared__ float sc1[LAYER], sc0[LAYER], coef[64], sb3[HROWS_PAD];
  int bb0 = blockIdx.x * 64;

  for (int j = threadIdx.x; j < 128; j += 192) {
    float c1 = g2[idx * LAYER + j] * rsqrtf(fs[2 * j + 1] + 1e-5f);
    sc1[j] = c1;
    sc0[j] = be2[idx * LAYER + j] - c1 * fs[2 * j];
  }
  for (int c = threadIdx.x; c < 64; c += 192) {
    int b = bb0 + c;
    coef[c] = dfm1 * S_old[b] * dif[b] * (SQDT_F * nrm[b]);
  }
  for (int r = threadIdx.x; r < HROWS_PAD; r += 192)
    sb3[r] = (r < HROWS) ? b3h[idx * HROWS + r] : 0.f;
  __syncthreads();

  // stage lrelu(BN(y2)) tile as f16
  for (int c = threadIdx.x; c < 1024; c += 192) {
    int row = c >> 4, j0 = (c & 15) * 8;
    const float* src = y2 + (size_t)(bb0 + row) * LAYER + j0;
    v8h o;
#pragma unroll
    for (int e = 0; e < 8; ++e)
      o[e] = (_Float16)lrelu(sc1[j0 + e] * src[e] + sc0[j0 + e]);
    *(v8h*)(tile + row * TP + j0) = o;
  }
  __syncthreads();

  int wave = threadIdx.x >> 5;                  // 0..5 -> m-tile
  int lane = threadIdx.x & 31;
  int hi   = lane >> 4;
  int m0   = wave * 16;
  int m    = m0 + (lane & 15);
  int aoff = hi * 8;

  v16h A[4];
#pragma unroll
  for (int kc = 0; kc < 4; ++kc) A[kc] = load_afrag(W, m, kc, aoff);

  int n = lane & 15;
  int koff = hi * 16;

#pragma unroll
  for (int nt = 0; nt < 4; ++nt) {
    v8f acc = {};
#pragma unroll
    for (int kc = 0; kc < 4; ++kc) {
      v16h Bf = *(const v16h*)(tile + (nt * 16 + n) * TP + kc * 32 + koff);
      acc = __builtin_amdgcn_wmma_f32_16x16x32_f16(false, A[kc], false, Bf,
                                                   (short)0, acc, false, false);
    }
    int bcol = bb0 + nt * 16 + n;
    float cf = coef[nt * 16 + n];
#pragma unroll
    for (int r = 0; r < 8; ++r) {
      int f = m0 + hi * 8 + r;
      if (f < HROWS) {
        size_t o = (size_t)f * B_PATHS + bcol;
        hedging[o] += cf * (acc[r] + sb3[f]);
      }
    }
  }
}

// --- maturity payoff: prices/variance (ddof=1) per strike, deterministic reduce
__global__ __launch_bounds__(256) void k_payoff(const float* __restrict__ strikes,
                                                const float* __restrict__ S_new,
                                                const float* __restrict__ hedging,
                                                int col, float dfi, float* __restrict__ out) {
  int k = blockIdx.x;
  float K = strikes[k];
  const float* hrow = hedging + (size_t)(k * N_MAT + col) * B_PATHS;
  __shared__ float ss[256], sq[256];
  float s = 0.f, q = 0.f;
  for (int b = threadIdx.x; b < B_PATHS; b += 256) {
    float p = dfi * fmaxf(S_new[b] - K, 0.f) - hrow[b];
    s += p; q += p * p;
  }
  ss[threadIdx.x] = s; sq[threadIdx.x] = q;
  __syncthreads();
  for (int o = 128; o > 0; o >>= 1) {
    if ((int)threadIdx.x < o) { ss[threadIdx.x] += ss[threadIdx.x + o]; sq[threadIdx.x] += sq[threadIdx.x + o]; }
    __syncthreads();
  }
  if (threadIdx.x == 0) {
    float sum = ss[0];
    out[k * N_MAT + col] = sum / B_PATHS;                                         // prices
    out[HROWS + k * N_MAT + col] = (sq[0] - sum * sum / B_PATHS) / (B_PATHS - 1); // variance
  }
}

// ---------------------------------------------------------------------------
extern "C" void kernel_launch(void* const* d_in, const int* in_sizes, int n_in,
                              void* d_out, int out_size, void* d_ws, size_t ws_size,
                              hipStream_t stream) {
  (void)in_sizes; (void)n_in; (void)out_size; (void)ws_size;
  // Input order: strikes, normals, maturity_steps, then lev/hedge param dicts
  // flattened in sorted-key (jax pytree) order: W1,W2,W3,b1,b2,b3,be1,be2,g1,g2.
  const float* strikes = (const float*)d_in[0];
  const float* normals = (const float*)d_in[1];
  // d_in[2] = maturity_steps (8,16,24,32): known at compile time.
  const float* lW1 = (const float*)d_in[3];
  const float* lW2 = (const float*)d_in[4];
  const float* lW3 = (const float*)d_in[5];
  // b1/b2 have zero effect (BatchNorm removes batch-constant terms) — unused.
  const float* lb3  = (const float*)d_in[8];
  const float* lbe1 = (const float*)d_in[9];
  const float* lbe2 = (const float*)d_in[10];
  const float* lg1  = (const float*)d_in[11];
  const float* lg2  = (const float*)d_in[12];
  const float* hW1 = (const float*)d_in[13];
  const float* hW2 = (const float*)d_in[14];
  const float* hW3 = (const float*)d_in[15];
  const float* hb3  = (const float*)d_in[18];
  const float* hbe1 = (const float*)d_in[19];
  const float* hbe2 = (const float*)d_in[20];
  const float* hg1  = (const float*)d_in[21];
  const float* hg2  = (const float*)d_in[22];
  float* out = (float*)d_out;

  // Workspace carve-up (256B aligned)
  char* ws = (char*)d_ws;
  size_t off = 0;
  auto take = [&](size_t bytes) -> char* {
    char* p = ws + off;
    off = (off + bytes + 255) & ~(size_t)255;
    return p;
  };
  _Float16* w2l16  = (_Float16*)take((size_t)N_MAT * LAYER * LAYER * 2);
  _Float16* w2h16  = (_Float16*)take((size_t)N_MAT * LAYER * LAYER * 2);
  _Float16* w3h16  = (_Float16*)take((size_t)N_MAT * HROWS_PAD * LAYER * 2);
  float* S_a     = (float*)take((size_t)B_PATHS * 4);
  float* S_b     = (float*)take((size_t)B_PATHS * 4);
  float* diffb   = (float*)take((size_t)B_PATHS * 4);
  float* statsS  = (float*)take(256);
  float* fstats  = (float*)take(2 * LAYER * 4);
  float* partials= (float*)take((size_t)NBLK1 * 256 * 4);
  float* y2      = (float*)take((size_t)B_PATHS * LAYER * 4);
  float* hedging = (float*)take((size_t)HROWS * B_PATHS * 4);

  const double DTd = 1.0 / 32.0;
  const float DF = (float)exp(-0.05 * DTd);

  k_convert<<<256, 256, 0, stream>>>(lW2, hW2, hW3, w2l16, w2h16, w3h16);
  k_init<<<(HROWS * B_PATHS + 255) / 256, 256, 0, stream>>>(hedging, S_a);

  float* Sbuf[2] = {S_a, S_b};

  for (int i = 1; i <= 32; ++i) {
    int idx = (i - 1) / 8;
    float dfm1 = powf(DF, (float)(i - 1));
    float dfi  = powf(DF, (float)i);
    float* S_old = Sbuf[(i - 1) & 1];
    float* S_new = Sbuf[i & 1];
    const float* nrm = normals + (size_t)(i - 1) * B_PATHS;

    k_stats_S<<<1, 1024, 0, stream>>>(S_old, statsS);

    // ---- lev net ----
    k_gemm1_fused<<<NBLK1, 256, 0, stream>>>(w2l16 + (size_t)idx * LAYER * LAYER,
                                             lW1, lg1, lbe1, idx, S_old, statsS, y2, partials);
    k_statsfin<<<128, 128, 0, stream>>>(partials, fstats);
    k_lev_final<<<B_PATHS / 256, 256, 0, stream>>>(y2, fstats, lg2, lbe2, lW3, lb3, idx,
                                                   S_old, nrm, S_new, diffb);

    // ---- hedge net (input uses pre-update S) ----
    k_gemm1_fused<<<NBLK1, 256, 0, stream>>>(w2h16 + (size_t)idx * LAYER * LAYER,
                                             hW1, hg1, hbe1, idx, S_old, statsS, y2, partials);
    k_statsfin<<<128, 128, 0, stream>>>(partials, fstats);
    k_gemm96_acc<<<B_PATHS / 64, 192, 0, stream>>>(w3h16 + (size_t)idx * HROWS_PAD * LAYER,
                                                   y2, fstats, hg2, hbe2, hb3, idx,
                                                   S_old, diffb, nrm, dfm1, hedging);

    if (i % 8 == 0)
      k_payoff<<<N_STRIKES, 256, 0, stream>>>(strikes, S_new, hedging, i / 8 - 1, dfi, out);
  }
}
